// SelfAttention_15479062135519
// MI455X (gfx1250) — compile-verified, matrix operand order checked by reading, not run on hardware
//
#include <hip/hip_runtime.h>

// Self-attention forward: Q=xWq*scale, K=xWk, V=xWv; O = softmax(QK^T)V
// B=4, T=4096, D_IN=768, D_H=256.  All matmuls via v_wmma_f32_16x16x32_f16.
// Attention kernel stages K/V chunks into LDS with global_load_async_to_lds_b128
// (double-buffered, ASYNCcnt-tracked) so the DMA overlaps the WMMA work.

typedef _Float16 half8 __attribute__((ext_vector_type(8)));
typedef _Float16 v16h  __attribute__((ext_vector_type(16)));
typedef float    v8f   __attribute__((ext_vector_type(8)));
typedef float    f32x4 __attribute__((ext_vector_type(4)));

#define B_  4
#define T_  4096
#define DIN 768
#define DH  256

// LDS row strides (halves), padded to spread banks; all row offsets 16B-aligned.
#define KSTR 264   // 32 K-rows   x 264 (528 B/row)
#define VSTR 40    // 256 V-rows  x 40  (80 B/row)
#define PSTR 40    // 16 P-rows   x 40  (80 B/row)

union Frag16 { v16h v; half8 h[2]; _Float16 e[16]; };

// 16x32 (A) or 32x16 (B) f16 fragment, per CDNA5 ISA 7.12.2 lane layout.
// lane<16 : row = lane,    K halves {0..7} and {16..23}
// lane>=16: row = lane-16, K halves {8..15} and {24..31}
__device__ __forceinline__ v16h load_frag_f16(const _Float16* base, int stride) {
  int lane = threadIdx.x & 31;
  int r    = lane & 15;
  int hi   = lane >> 4;
  const _Float16* p = base + (size_t)r * stride + (hi ? 8 : 0);
  Frag16 f;
  f.h[0] = *(const half8*)(p);
  f.h[1] = *(const half8*)(p + 16);
  return f.v;
}

// Same A-fragment layout but source is fp32 (converted to f16 in registers).
__device__ __forceinline__ v16h load_frag_f32cvt(const float* base, int stride) {
  int lane = threadIdx.x & 31;
  int r    = lane & 15;
  int hi   = lane >> 4;
  const float* p = base + (size_t)r * stride + (hi ? 8 : 0);
  f32x4 f0 = *(const f32x4*)(p);
  f32x4 f1 = *(const f32x4*)(p + 4);
  f32x4 f2 = *(const f32x4*)(p + 16);
  f32x4 f3 = *(const f32x4*)(p + 20);
  Frag16 f;
#pragma unroll
  for (int i = 0; i < 4; ++i) {
    f.e[i]      = (_Float16)f0[i];
    f.e[4 + i]  = (_Float16)f1[i];
    f.e[8 + i]  = (_Float16)f2[i];
    f.e[12 + i] = (_Float16)f3[i];
  }
  return f.v;
}

__device__ __forceinline__ v8f wmma_f16(v16h a, v16h b, v8f c) {
  // (neg_a, A, neg_b, B, c_mod, C, reuse_a, reuse_b)
  return __builtin_amdgcn_wmma_f32_16x16x32_f16(false, a, false, b, (short)0, c,
                                                false, false);
}

// Async-DMA one 16B segment: LDS[dst] = MEM[saddr + voff]  (GVS mode, ASYNCcnt)
__device__ __forceinline__ void async_b128(unsigned lds_dst, unsigned voff,
                                           const _Float16* saddr) {
  __asm__ volatile("global_load_async_to_lds_b128 %0, %1, %2"
                   :: "v"(lds_dst), "v"(voff), "s"(saddr) : "memory");
}

// ---------------------------------------------------------------------------
// Kernel 0: transpose+convert weights: Wt[w][h][d] = (f16) W_w[d][h]
// ---------------------------------------------------------------------------
__global__ void wt_kernel(const float* __restrict__ Wq, const float* __restrict__ Wk,
                          const float* __restrict__ Wv, _Float16* __restrict__ Wt) {
  int idx = blockIdx.x * blockDim.x + threadIdx.x;
  const int per = DIN * DH;
  if (idx >= 3 * per) return;
  int w   = idx / per;
  int rem = idx - w * per;
  int k   = rem / DH;          // d index
  int n   = rem - k * DH;      // h index
  const float* src = (w == 0) ? Wq : (w == 1) ? Wk : Wv;
  Wt[(size_t)w * per + (size_t)n * DIN + k] = (_Float16)src[rem];
}

// ---------------------------------------------------------------------------
// Kernel 1: QKV projection GEMM.  grid=(128,3), block=256 (8 waves).
// Each wave: 16 rows x full DH=256 cols.  blockIdx.y selects Q/K/V.
// Q is pre-scaled by 1/16; V stored transposed Vt[b][h][t].
// ---------------------------------------------------------------------------
__global__ __launch_bounds__(256) void qkv_kernel(
    const float* __restrict__ x, const _Float16* __restrict__ Wt,
    _Float16* __restrict__ Qh, _Float16* __restrict__ Kh,
    _Float16* __restrict__ Vt) {
  int w       = threadIdx.x >> 5;
  int lane    = threadIdx.x & 31;
  int wsel    = blockIdx.y;
  int rowbase = blockIdx.x * 128 + w * 16;   // global row in [0, B*T)
  const _Float16* Wsel = Wt + (size_t)wsel * (DIN * DH);

  v8f zv = {};
  v8f acc[16];
#pragma unroll
  for (int n = 0; n < 16; ++n) acc[n] = zv;

  const float* xbase = x + (size_t)rowbase * DIN;
  for (int c = 0; c < DIN / 32; ++c) {
    if (c + 1 < DIN / 32)
      __builtin_prefetch(xbase + (size_t)(lane & 15) * DIN + (c + 1) * 32, 0, 3);
    v16h af = load_frag_f32cvt(xbase + c * 32, DIN);
#pragma unroll
    for (int n = 0; n < 16; ++n) {
      v16h bf = load_frag_f16(Wsel + (size_t)(n * 16) * DIN + c * 32, DIN);
      acc[n] = wmma_f16(af, bf, acc[n]);
    }
  }

  float scale = (wsel == 0) ? 0.0625f : 1.0f;   // fold 1/sqrt(DH) into Q
  int nlo  = lane & 15;
  int mofs = (lane >> 4) * 8;
#pragma unroll
  for (int n = 0; n < 16; ++n) {
#pragma unroll
    for (int r = 0; r < 8; ++r) {
      int m    = r + mofs;
      int grow = rowbase + m;
      int col  = n * 16 + nlo;
      _Float16 hv = (_Float16)(acc[n][r] * scale);
      if (wsel == 0) {
        Qh[(size_t)grow * DH + col] = hv;
      } else if (wsel == 1) {
        Kh[(size_t)grow * DH + col] = hv;
      } else {
        int b = grow >> 12;              // T_ == 4096
        int t = grow & (T_ - 1);
        Vt[((size_t)b * DH + col) * T_ + t] = hv;
      }
    }
  }
}

// ---------------------------------------------------------------------------
// Kernel 2: flash attention.  grid=(T/128, B), block=256 (8 waves).
// Each wave: 16 queries, full DH output, online softmax over keys.
// K/V chunks (32 keys) staged to LDS by async DMA, double-buffered.
// ---------------------------------------------------------------------------
__global__ __launch_bounds__(256) void attn_kernel(
    const _Float16* __restrict__ Qh, const _Float16* __restrict__ Kh,
    const _Float16* __restrict__ Vt, float* __restrict__ out) {
  __shared__ __align__(16) _Float16 kbuf[2][32 * KSTR];   // K chunk:  32 x 256
  __shared__ __align__(16) _Float16 vbuf[2][256 * VSTR];  // V^T chunk: 256 x 32
  __shared__ __align__(16) _Float16 pbuf[8][16 * PSTR];   // per-wave P tile

  int tid   = threadIdx.x;
  int w     = tid >> 5;
  int lane  = tid & 31;
  int b     = blockIdx.y;
  int qbase = blockIdx.x * 128 + w * 16;
  int nlo   = lane & 15;
  int mofs  = (lane >> 4) * 8;
  _Float16* pb = &pbuf[w][0];

  unsigned klds[2] = { (unsigned)(uintptr_t)&kbuf[0][0],
                       (unsigned)(uintptr_t)&kbuf[1][0] };
  unsigned vlds[2] = { (unsigned)(uintptr_t)&vbuf[0][0],
                       (unsigned)(uintptr_t)&vbuf[1][0] };

  // Q resident in registers: 8 A-fragments covering DH=256.
  v16h aq[8];
  const _Float16* qptr = Qh + (size_t)(b * T_ + qbase) * DH;
#pragma unroll
  for (int c = 0; c < 8; ++c) aq[c] = load_frag_f16(qptr + c * 32, DH);

  v8f zv = {};
  v8f o[16];
#pragma unroll
  for (int t = 0; t < 16; ++t) o[t] = zv;

  float mstate[8], lstate[8];
#pragma unroll
  for (int r = 0; r < 8; ++r) { mstate[r] = -1e30f; lstate[r] = 0.0f; }

  // --- async-copy of one 32-key chunk (K: 16KB, V^T: 16KB), 8 b128/thread ---
  auto issue_chunk = [&](int buf, int kb) {
    const _Float16* kg = Kh + (size_t)(b * T_ + kb) * DH;   // 32 rows x 512B
    const _Float16* vg = Vt + (size_t)b * DH * T_ + kb;     // 256 rows x 64B
#pragma unroll
    for (int it = 0; it < 4; ++it) {
      int s = tid + it * 256;             // 1024 segments of 16B
      int row = s >> 5, c = s & 31;
      async_b128(klds[buf] + row * (KSTR * 2) + c * 16,
                 (unsigned)(row * (DH * 2) + c * 16), kg);
    }
#pragma unroll
    for (int it = 0; it < 4; ++it) {
      int s = tid + it * 256;             // 1024 segments of 16B
      int row = s >> 2, j = s & 3;
      async_b128(vlds[buf] + row * (VSTR * 2) + j * 16,
                 (unsigned)(row * (T_ * 2) + j * 16), vg);
    }
  };

  issue_chunk(0, 0);   // prologue

  for (int kb = 0, it = 0; kb < T_; kb += 32, ++it) {
    int cur = it & 1;
    __asm__ volatile("s_wait_asynccnt 0" ::: "memory");  // chunk landed in LDS
    __syncthreads();                                     // publish; prev reads done
    if (kb + 32 < T_) issue_chunk(cur ^ 1, kb + 32);     // overlap next DMA

    const _Float16* kl = &kbuf[cur][0];
    const _Float16* vl = &vbuf[cur][0];

    // --- S = Q * K^T for 32 keys (two 16x16 tiles) ---
    v8f s0 = zv, s1 = zv;
#pragma unroll
    for (int c = 0; c < 8; ++c) {
      v16h bk0 = load_frag_f16(kl + c * 32, KSTR);
      v16h bk1 = load_frag_f16(kl + 16 * KSTR + c * 32, KSTR);
      s0 = wmma_f16(aq[c], bk0, s0);
      s1 = wmma_f16(aq[c], bk1, s1);
    }

    // --- online softmax; rows live across the 16-lane half (C layout) ---
    float alpha[8];
#pragma unroll
    for (int r = 0; r < 8; ++r) {
      float x0 = s0[r], x1 = s1[r];
      float mx = fmaxf(x0, x1);
      mx = fmaxf(mx, __shfl_xor(mx, 1));
      mx = fmaxf(mx, __shfl_xor(mx, 2));
      mx = fmaxf(mx, __shfl_xor(mx, 4));
      mx = fmaxf(mx, __shfl_xor(mx, 8));
      float mnew = fmaxf(mstate[r], mx);
      float a  = __expf(mstate[r] - mnew);
      float p0 = __expf(x0 - mnew);
      float p1 = __expf(x1 - mnew);
      float rs = p0 + p1;
      rs += __shfl_xor(rs, 1);
      rs += __shfl_xor(rs, 2);
      rs += __shfl_xor(rs, 4);
      rs += __shfl_xor(rs, 8);
      lstate[r] = lstate[r] * a + rs;
      mstate[r] = mnew;
      alpha[r]  = a;
      int m = r + mofs;                    // C layout -> (row m, key nlo)
      pb[m * PSTR + nlo]      = (_Float16)p0;
      pb[m * PSTR + 16 + nlo] = (_Float16)p1;
    }
#pragma unroll
    for (int t = 0; t < 16; ++t)
#pragma unroll
      for (int r = 0; r < 8; ++r) o[t][r] *= alpha[r];

    __asm__ volatile("s_wait_dscnt 0" ::: "memory");  // P tile visible in LDS

    // --- O += P * V : P as A-fragment (K=32), V^T rows from LDS ---
    v16h pa = load_frag_f16(pb, PSTR);
#pragma unroll
    for (int t = 0; t < 16; ++t) {
      v16h bv = load_frag_f16(vl + (t * 16) * VSTR, VSTR);
      o[t] = wmma_f16(pa, bv, o[t]);
    }
  }

  // --- normalize and write fp32 output ---
  float inv[8];
#pragma unroll
  for (int r = 0; r < 8; ++r) inv[r] = 1.0f / lstate[r];
#pragma unroll
  for (int t = 0; t < 16; ++t) {
#pragma unroll
    for (int r = 0; r < 8; ++r) {
      int m = r + mofs;
      int q = qbase + m;
      out[(size_t)(b * T_ + q) * DH + t * 16 + nlo] = o[t][r] * inv[r];
    }
  }
}

// ---------------------------------------------------------------------------
extern "C" void kernel_launch(void* const* d_in, const int* in_sizes, int n_in,
                              void* d_out, int out_size, void* d_ws, size_t ws_size,
                              hipStream_t stream) {
  const float* x  = (const float*)d_in[0];
  const float* Wq = (const float*)d_in[1];
  const float* Wk = (const float*)d_in[2];
  const float* Wv = (const float*)d_in[3];
  float* out = (float*)d_out;

  char* ws = (char*)d_ws;
  _Float16* Qh = (_Float16*)(ws);                                 // 8 MB
  _Float16* Kh = (_Float16*)(ws + (size_t)8  * 1024 * 1024);      // 8 MB
  _Float16* Vt = (_Float16*)(ws + (size_t)16 * 1024 * 1024);      // 8 MB
  _Float16* Wt = (_Float16*)(ws + (size_t)24 * 1024 * 1024);      // 3 x 384 KB

  int total_w = 3 * DIN * DH;
  wt_kernel<<<(total_w + 255) / 256, 256, 0, stream>>>(Wq, Wk, Wv, Wt);

  dim3 g1((B_ * T_) / 128, 3);
  qkv_kernel<<<g1, 256, 0, stream>>>(x, Wt, Qh, Kh, Vt);

  dim3 g2(T_ / 128, B_);
  attn_kernel<<<g2, 256, 0, stream>>>(Qh, Kh, Vt, out);
}
